// MultiDecoderSeq2SeqAttn_44504451121244
// MI455X (gfx1250) — matrix operand order repaired and checked by reading, not compiled, not run on hardware
//
#include <hip/hip_runtime.h>

// ---------------- model dims ----------------
static constexpr int kH    = 1024;     // hidden
static constexpr int kL    = 512;      // encoder length
static constexpr int kG4   = 4096;     // 4*H gate rows
static constexpr int kVOUT = 32000;    // output vocab
static constexpr int kSTART = 1, kSTOP = 2, kPAD = 0;

typedef __attribute__((ext_vector_type(16))) __bf16 v16bf;
typedef __attribute__((ext_vector_type(8)))  float  v8f;
typedef unsigned short u16;
typedef unsigned int   u32;

// ---------------- device helpers ----------------
__device__ __forceinline__ float bflo(u32 u) { return __uint_as_float(u << 16); }
__device__ __forceinline__ float bfhi(u32 u) { return __uint_as_float(u & 0xffff0000u); }
__device__ __forceinline__ u16 f2bf(float f) {          // round-to-nearest-even f32 -> bf16
    u32 u = __float_as_uint(f);
    u32 r = u + 0x7fffu + ((u >> 16) & 1u);
    return (u16)(r >> 16);
}
__device__ __forceinline__ float sigm(float x) { return 1.0f / (1.0f + __expf(-x)); }

// Device-scope grid barrier for a persistent kernel (all blocks resident).
__device__ void gsync(int* cnt, volatile int* gen, int nblk) {
    __threadfence();
    __syncthreads();
    if (threadIdx.x == 0) {
        int g = *gen;
        if (atomicAdd(cnt, 1) == nblk - 1) {
            atomicExch(cnt, 0);
            __threadfence();
            atomicAdd((int*)gen, 1);
        } else {
            while (*gen == g) { __builtin_amdgcn_s_sleep(2); }
        }
    }
    __syncthreads();
    __threadfence();
}

// ---------------- kernels ----------------
__global__ void k_init(int* state, int* out, int* scnt, int* sgen) {
    if (threadIdx.x == 0) {
        state[0] = kSTART;   // current token
        state[1] = 0;        // done flag
        out[0]   = kSTART;
        *scnt = 0; *sgen = 0;
    }
}

__global__ void k_cvt(const float* __restrict__ in, u16* __restrict__ out, int n) {
    int i = blockIdx.x * blockDim.x + threadIdx.x;
    int stride = gridDim.x * blockDim.x;
    for (; i < n; i += stride) out[i] = f2bf(in[i]);
}

__global__ void k_gather_embed(const float* __restrict__ embed, const int* __restrict__ toks,
                               u16* __restrict__ out) {
    int t = blockIdx.x;
    const float* src = embed + (size_t)toks[t] * kH;
    for (int i = threadIdx.x; i < kH; i += blockDim.x)
        out[(size_t)t * kH + i] = f2bf(src[i]);
}

// C[M,N] = A[M,K] * W[N,K]^T + bias[n]   (bf16 in, f32 accum) via V_WMMA_F32_16X16X32_BF16.
// Register-blocked: each wave computes a 16(M) x 64(N) strip = 4 accumulators sharing one
// A fragment per K-chunk (4 independent back-to-back wmmas hide the WMMA RAW latency).
// grid (N/256, M/16), block 128 (4 waves).
__global__ void k_gemm_wmma(const u16* __restrict__ A, const u16* __restrict__ W,
                            const float* __restrict__ bias, float* __restrict__ C,
                            int M, int N, int K) {
    int lane = threadIdx.x & 31, wave = threadIdx.x >> 5;
    int tm  = blockIdx.y * 16;
    int tn0 = blockIdx.x * 256 + wave * 64;
    int lm = lane & 15, lh = lane >> 4;

    const u32* Ap = (const u32*)(A + (size_t)(tm + lm) * K);    // A row for M = tm+lm
    const u32* Wp = (const u32*)(W + (size_t)(tn0 + lm) * K);   // W row for N = tn0+lm (tile 0)
    const int  rowStride = K >> 1;                              // uints per W row
    v8f acc[4];
    #pragma unroll
    for (int nt = 0; nt < 4; ++nt) acc[nt] = (v8f){0.f,0.f,0.f,0.f,0.f,0.f,0.f,0.f};

    for (int kk = 0; kk < K; kk += 32) {
        int kb = kk >> 1;                                       // uint (2xbf16) index base
        union { v16bf v; u32 u[8]; } a;
        #pragma unroll
        for (int vv = 0; vv < 8; ++vv)
            // A 16x32 layout: klocal = (vv>=4)*16 + lh*8 + (vv%4)*2 + e -> uint idx
            a.u[vv] = Ap[kb + ((vv >= 4) ? 8 : 0) + lh * 4 + (vv & 3)];
        #pragma unroll
        for (int nt = 0; nt < 4; ++nt) {
            union { v16bf v; u32 u[8]; } b;
            const u32* Wt = Wp + (size_t)nt * 16 * rowStride;   // rows for N-tile nt
            #pragma unroll
            for (int vv = 0; vv < 8; ++vv)
                // B 32x16 layout: klocal = (vv>=4)*16 + lh*8 + (vv%4)*2 + e -> uint idx
                b.u[vv] = Wt[kb + ((vv >= 4) ? 8 : 0) + lh * 4 + (vv & 3)];
            acc[nt] = __builtin_amdgcn_wmma_f32_16x16x32_bf16(false, a.v, false, b.v,
                                                              (short)0, acc[nt], false, false);
        }
    }
    #pragma unroll
    for (int nt = 0; nt < 4; ++nt) {
        #pragma unroll
        for (int r = 0; r < 8; ++r) {
            int m = tm + lh * 8 + r;
            int n = tn0 + nt * 16 + lm;
            C[(size_t)m * N + n] = acc[nt][r] + bias[n];
        }
    }
}

// Persistent LSTM layer: all kL steps in one launch, grid barrier between steps.
// 64 blocks x 256; block owns 16 hidden units (64 gate rows); wave handles 8 rows.
// h double-buffered (read hin, write hout); c updated in place (owner-only).
__global__ void k_enc_layer(const u16* __restrict__ Whh, const float* __restrict__ bhh,
                            const float* __restrict__ XP, float* __restrict__ hA,
                            float* __restrict__ hB, float* __restrict__ c,
                            u16* __restrict__ obf, float* __restrict__ of32,
                            int* scnt, int* sgen) {
    __shared__ float gl[4][16];
    int lane = threadIdx.x & 31, wave = threadIdx.x >> 5;

    if (threadIdx.x < 16) { int u = blockIdx.x * 16 + threadIdx.x; hA[u] = 0.f; c[u] = 0.f; }
    gsync(scnt, sgen, gridDim.x);

    for (int t = 0; t < kL; ++t) {
        const float*  xprow = XP + (size_t)t * kG4;
        const float2* hp    = (const float2*)((t & 1) ? hB : hA);
        float*        hout  = (t & 1) ? hA : hB;

        for (int j = 0; j < 8; ++j) {
            int lin = wave * 8 + j;
            int ul = lin & 15, g = lin >> 4;
            int u = blockIdx.x * 16 + ul;
            int row = g * kH + u;
            const u32* wp = (const u32*)(Whh + (size_t)row * kH);
            float acc = 0.f;
            for (int k = lane; k < kH / 2; k += 32) {
                u32 w = wp[k]; float2 hv = hp[k];
                acc = fmaf(bflo(w), hv.x, acc);
                acc = fmaf(bfhi(w), hv.y, acc);
            }
            #pragma unroll
            for (int off = 16; off; off >>= 1) acc += __shfl_down(acc, off, 32);
            if (lane == 0) gl[g][ul] = acc + xprow[row] + bhh[row];
        }
        __syncthreads();
        if (threadIdx.x < 16) {
            int ul = threadIdx.x, u = blockIdx.x * 16 + ul;
            float gi = gl[0][ul], gf = gl[1][ul], gg = gl[2][ul], go = gl[3][ul];
            float cc = sigm(gf) * c[u] + sigm(gi) * tanhf(gg);
            float hh = sigm(go) * tanhf(cc);
            c[u] = cc;
            hout[u] = hh;
            if (obf)  obf[(size_t)t * kH + u]  = f2bf(hh);
            if (of32) of32[(size_t)t * kH + u] = hh;
        }
        gsync(scnt, sgen, gridDim.x);   // kL even => final h lands in hA
    }
}

// Wave-per-row bf16 mat-vec: y[row] = dot(W[row,:], x) + bias[row], optional relu.
__global__ void k_matvec(const u16* __restrict__ W, const float* __restrict__ bias,
                         const float* __restrict__ x, float* __restrict__ y,
                         int rows, int K, int relu) {
    int lane = threadIdx.x & 31, wave = threadIdx.x >> 5;
    int row = blockIdx.x * 8 + wave;
    if (row >= rows) return;
    if (row + 8 < rows) __builtin_prefetch(W + (size_t)(row + 8) * K, 0, 1);
    const u32*    wp = (const u32*)(W + (size_t)row * K);
    const float2* xp = (const float2*)x;
    int K2 = K >> 1;
    float acc = 0.f;
    for (int k = lane; k < K2; k += 32) {
        u32 w = wp[k]; float2 xv = xp[k];
        acc = fmaf(bflo(w), xv.x, acc);
        acc = fmaf(bfhi(w), xv.y, acc);
    }
    #pragma unroll
    for (int off = 16; off; off >>= 1) acc += __shfl_down(acc, off, 32);
    if (lane == 0) {
        float v = acc + bias[row];
        if (relu) v = fmaxf(v, 0.f);
        y[row] = v;
    }
}

__global__ void k_softmax512(const float* __restrict__ x, float* __restrict__ y) {
    __shared__ float red[512];
    int tid = threadIdx.x;
    float v = x[tid];
    red[tid] = v; __syncthreads();
    for (int s = 256; s; s >>= 1) { if (tid < s) red[tid] = fmaxf(red[tid], red[tid + s]); __syncthreads(); }
    float mx = red[0]; __syncthreads();
    float e = __expf(v - mx);
    red[tid] = e; __syncthreads();
    for (int s = 256; s; s >>= 1) { if (tid < s) red[tid] += red[tid + s]; __syncthreads(); }
    y[tid] = e / red[0];
}

__global__ void k_ctx(const float* __restrict__ aw, const float* __restrict__ eo,
                      float* __restrict__ ctx_out) {
    int j = blockIdx.x * blockDim.x + threadIdx.x;  // 0..kH-1
    float acc = 0.f;
    for (int t = 0; t < kL; ++t) acc = fmaf(aw[t], eo[(size_t)t * kH + j], acc);
    ctx_out[j] = acc;
}

__global__ void k_prepare(const float* __restrict__ dec_embed, const int* __restrict__ state,
                          const float* __restrict__ h, float* __restrict__ cat1,
                          float* __restrict__ cat2) {
    int i = blockIdx.x * blockDim.x + threadIdx.x;  // 0..kH-1
    int tok = state[0];
    float e = dec_embed[(size_t)tok * kH + i];
    cat1[i] = e;
    cat2[i] = e;
    cat1[kH + i] = h[i];
}

// Decoder LSTM cell: candidate (h2,c2) always computed; commit happens in k_finalize.
__global__ void k_dec_lstm(const u16* __restrict__ Wih, const u16* __restrict__ Whh,
                           const float* __restrict__ bih, const float* __restrict__ bhh,
                           const float* __restrict__ x, const float* __restrict__ h,
                           const float* __restrict__ c, float* __restrict__ h2,
                           float* __restrict__ c2) {
    __shared__ float gl[4][16];
    int lane = threadIdx.x & 31, wave = threadIdx.x >> 5;
    const float2* xp = (const float2*)x;
    const float2* hp = (const float2*)h;
    for (int j = 0; j < 8; ++j) {
        int lin = wave * 8 + j;
        int ul = lin & 15, g = lin >> 4;
        int u = blockIdx.x * 16 + ul;
        int row = g * kH + u;
        const u32* wi = (const u32*)(Wih + (size_t)row * kH);
        const u32* wh = (const u32*)(Whh + (size_t)row * kH);
        float acc = 0.f;
        for (int k = lane; k < kH / 2; k += 32) {
            u32 a = wi[k]; float2 xv = xp[k];
            acc = fmaf(bflo(a), xv.x, acc);
            acc = fmaf(bfhi(a), xv.y, acc);
            u32 b = wh[k]; float2 hv = hp[k];
            acc = fmaf(bflo(b), hv.x, acc);
            acc = fmaf(bfhi(b), hv.y, acc);
        }
        #pragma unroll
        for (int off = 16; off; off >>= 1) acc += __shfl_down(acc, off, 32);
        if (lane == 0) gl[g][ul] = acc + bih[row] + bhh[row];
    }
    __syncthreads();
    if (threadIdx.x < 16) {
        int ul = threadIdx.x, u = blockIdx.x * 16 + ul;
        float gi = gl[0][ul], gf = gl[1][ul], gg = gl[2][ul], go = gl[3][ul];
        float cc = sigm(gf) * c[u] + sigm(gi) * tanhf(gg);
        h2[u] = sigm(go) * tanhf(cc);
        c2[u] = cc;
    }
}

__global__ void k_argmax1(const float* __restrict__ logits, float* __restrict__ pval,
                          int* __restrict__ pidx) {
    __shared__ float sv[256]; __shared__ int si[256];
    int tid = threadIdx.x;
    int i = blockIdx.x * 256 + tid;                  // 125*256 == 32000 exactly
    sv[tid] = logits[i]; si[tid] = i; __syncthreads();
    for (int s = 128; s; s >>= 1) {
        if (tid < s) {
            if (sv[tid + s] > sv[tid] || (sv[tid + s] == sv[tid] && si[tid + s] < si[tid])) {
                sv[tid] = sv[tid + s]; si[tid] = si[tid + s];
            }
        }
        __syncthreads();
    }
    if (tid == 0) { pval[blockIdx.x] = sv[0]; pidx[blockIdx.x] = si[0]; }
}

__global__ void k_finalize(const float* __restrict__ pval, const int* __restrict__ pidx,
                           int nparts, int* state, float* h, float* c,
                           const float* __restrict__ h2, const float* __restrict__ c2,
                           int* out, int step) {
    __shared__ float sv[256]; __shared__ int si[256]; __shared__ int sdone;
    int tid = threadIdx.x;
    float v = -3.4e38f; int idx = 0x7fffffff;
    if (tid < nparts) { v = pval[tid]; idx = pidx[tid]; }
    sv[tid] = v; si[tid] = idx; __syncthreads();
    for (int s = 128; s; s >>= 1) {
        if (tid < s) {
            if (sv[tid + s] > sv[tid] || (sv[tid + s] == sv[tid] && si[tid + s] < si[tid])) {
                sv[tid] = sv[tid + s]; si[tid] = si[tid + s];
            }
        }
        __syncthreads();
    }
    if (tid == 0) {
        int nxt = si[0];
        int doneOld = state[1];
        sdone = doneOld;
        out[1 + step] = doneOld ? kPAD : nxt;
        if (!doneOld) state[0] = nxt;
        state[1] = doneOld | (nxt == kSTOP);
    }
    __syncthreads();
    if (!sdone) {
        for (int i = tid; i < kH; i += blockDim.x) { h[i] = h2[i]; c[i] = c2[i]; }
    }
}

// ---------------- host ----------------
extern "C" void kernel_launch(void* const* d_in, const int* in_sizes, int n_in,
                              void* d_out, int out_size, void* d_ws, size_t ws_size,
                              hipStream_t stream) {
    (void)in_sizes; (void)n_in; (void)ws_size;
    const int*   enc_tokens = (const int*)d_in[0];
    const float* enc_embed  = (const float*)d_in[1];
    const float* enc1_Wih   = (const float*)d_in[2];
    const float* enc1_Whh   = (const float*)d_in[3];
    const float* enc1_bih   = (const float*)d_in[4];
    const float* enc1_bhh   = (const float*)d_in[5];
    const float* enc2_Wih   = (const float*)d_in[6];
    const float* enc2_Whh   = (const float*)d_in[7];
    const float* enc2_bih   = (const float*)d_in[8];
    const float* enc2_bhh   = (const float*)d_in[9];
    const float* dec_embed  = (const float*)d_in[10];
    const float* attn_Wf    = (const float*)d_in[11];
    const float* attn_bf    = (const float*)d_in[12];
    const float* comb_Wf    = (const float*)d_in[13];
    const float* comb_bf    = (const float*)d_in[14];
    const float* dec_Wih    = (const float*)d_in[15];
    const float* dec_Whh    = (const float*)d_in[16];
    const float* dec_bih    = (const float*)d_in[17];
    const float* dec_bhh    = (const float*)d_in[18];
    const float* out_Wf     = (const float*)d_in[19];
    const float* out_bf     = (const float*)d_in[20];

    char* p = (char*)d_ws;
    auto carve = [&](size_t bytes) -> char* {
        char* r = p; p += (bytes + 255) & ~(size_t)255; return r;
    };
    const size_t nGate = (size_t)kG4 * kH;          // 4096*1024
    u16* W1ih_b = (u16*)carve(2 * nGate);
    u16* W1hh_b = (u16*)carve(2 * nGate);
    u16* W2ih_b = (u16*)carve(2 * nGate);
    u16* W2hh_b = (u16*)carve(2 * nGate);
    u16* Wdih_b = (u16*)carve(2 * nGate);
    u16* Wdhh_b = (u16*)carve(2 * nGate);
    u16* Wcomb_b = (u16*)carve(2 * (size_t)kH * 2 * kH);
    u16* Wattn_b = (u16*)carve(2 * (size_t)kL * 2 * kH);
    u16* Wout_b  = (u16*)carve(2 * (size_t)kVOUT * kH);
    u16* embA    = (u16*)carve(2 * (size_t)kL * kH);
    u16* o1b     = (u16*)carve(2 * (size_t)kL * kH);
    float* XP       = (float*)carve(4 * (size_t)kL * kG4);
    float* enc_outs = (float*)carve(4 * (size_t)kL * kH);
    float* hA   = (float*)carve(4 * kH);
    float* hB   = (float*)carve(4 * kH);
    float* cbuf = (float*)carve(4 * kH);
    float* h2   = (float*)carve(4 * kH);
    float* c2   = (float*)carve(4 * kH);
    float* cat1 = (float*)carve(4 * 2 * kH);
    float* cat2 = (float*)carve(4 * 2 * kH);
    float* attn_logits = (float*)carve(4 * kL);
    float* attn_w      = (float*)carve(4 * kL);
    float* comb        = (float*)carve(4 * kH);
    float* logits      = (float*)carve(4 * kVOUT);
    float* pval = (float*)carve(4 * 128);
    int*   pidx = (int*)carve(4 * 128);
    int*   state = (int*)carve(64);
    int*   scnt  = (int*)carve(64);
    int*   sgen  = (int*)carve(64);

    k_init<<<1, 64, 0, stream>>>(state, (int*)d_out, scnt, sgen);

    // bf16 weight conversion (all fit in L2 afterwards: ~118 MB)
    k_cvt<<<1024, 256, 0, stream>>>(enc1_Wih, W1ih_b, (int)nGate);
    k_cvt<<<1024, 256, 0, stream>>>(enc1_Whh, W1hh_b, (int)nGate);
    k_cvt<<<1024, 256, 0, stream>>>(enc2_Wih, W2ih_b, (int)nGate);
    k_cvt<<<1024, 256, 0, stream>>>(enc2_Whh, W2hh_b, (int)nGate);
    k_cvt<<<1024, 256, 0, stream>>>(dec_Wih,  Wdih_b, (int)nGate);
    k_cvt<<<1024, 256, 0, stream>>>(dec_Whh,  Wdhh_b, (int)nGate);
    k_cvt<<<1024, 256, 0, stream>>>(comb_Wf,  Wcomb_b, kH * 2 * kH);
    k_cvt<<<1024, 256, 0, stream>>>(attn_Wf,  Wattn_b, kL * 2 * kH);
    k_cvt<<<1024, 256, 0, stream>>>(out_Wf,   Wout_b,  kVOUT * kH);

    // encoder
    k_gather_embed<<<kL, 256, 0, stream>>>(enc_embed, enc_tokens, embA);
    dim3 gg(kG4 / 256, kL / 16);    // register-blocked: wave = 16x64 strip
    k_gemm_wmma<<<gg, 128, 0, stream>>>(embA, W1ih_b, enc1_bih, XP, kL, kG4, kH);
    k_enc_layer<<<64, 256, 0, stream>>>(W1hh_b, enc1_bhh, XP, hA, hB, cbuf,
                                        o1b, nullptr, scnt, sgen);
    k_gemm_wmma<<<gg, 128, 0, stream>>>(o1b, W2ih_b, enc2_bih, XP, kL, kG4, kH);
    k_enc_layer<<<64, 256, 0, stream>>>(W2hh_b, enc2_bhh, XP, hA, hB, cbuf,
                                        nullptr, enc_outs, scnt, sgen);

    // decoder: fixed number of steps = out_size - 1 (reference masks with PAD after STOP)
    int steps = out_size - 1;
    for (int s = 0; s < steps; ++s) {
        k_prepare<<<4, 256, 0, stream>>>(dec_embed, state, hA, cat1, cat2);
        k_matvec<<<64, 256, 0, stream>>>(Wattn_b, attn_bf, cat1, attn_logits, kL, 2 * kH, 0);
        k_softmax512<<<1, 512, 0, stream>>>(attn_logits, attn_w);
        k_ctx<<<4, 256, 0, stream>>>(attn_w, enc_outs, cat2 + kH);
        k_matvec<<<128, 256, 0, stream>>>(Wcomb_b, comb_bf, cat2, comb, kH, 2 * kH, 1);
        k_dec_lstm<<<64, 256, 0, stream>>>(Wdih_b, Wdhh_b, dec_bih, dec_bhh,
                                           comb, hA, cbuf, h2, c2);
        k_matvec<<<kVOUT / 8, 256, 0, stream>>>(Wout_b, out_bf, h2, logits, kVOUT, kH, 0);
        k_argmax1<<<kVOUT / 256, 256, 0, stream>>>(logits, pval, pidx);
        k_finalize<<<1, 256, 0, stream>>>(pval, pidx, kVOUT / 256, state, hA, cbuf,
                                          h2, c2, (int*)d_out, s);
    }
}
